// GroupedQueryAttention_61761629717101
// MI455X (gfx1250) — matrix operand order repaired and checked by reading, not compiled, not run on hardware
//
#include <hip/hip_runtime.h>
#include <hip/hip_bf16.h>
#include <math.h>

#define H_ 16
#define KV_ 4
#define HD_ 128
#define REP_ (H_ / KV_)
#define B_ 4
#define T_ 2048
#define E_ 2048
#define MTOT (B_ * T_)

typedef __attribute__((ext_vector_type(16))) __bf16 v16bf;
typedef __attribute__((ext_vector_type(8))) float v8f;
typedef __attribute__((ext_vector_type(8))) unsigned short u16x8;
typedef __attribute__((ext_vector_type(4))) unsigned int u32x4;
typedef __attribute__((ext_vector_type(8))) int i32x8;

union FragU { u16x8 h[2]; v16bf v; };

__device__ __forceinline__ unsigned short f2bf(float f) {
  union { float f; unsigned int u; } x; x.f = f;
  unsigned int u = x.u;
  return (unsigned short)((u + 0x7fffu + ((u >> 16) & 1u)) >> 16);
}

__device__ __forceinline__ v8f wmma_bf16(v16bf a, v16bf b, v8f c) {
  return __builtin_amdgcn_wmma_f32_16x16x32_bf16(false, a, false, b, (short)0, c,
                                                 false, false);
}

// A fragment (16x32): row-major bf16 [rows][pitch]; per-lane row,
// K pattern {kc+kb+0..7, kc+kb+16..23}, kb = 8*(lane>=16).
__device__ __forceinline__ v16bf load_a_frag(const unsigned short* base, int pitch,
                                             int row0, int kc, int lane) {
  int m = row0 + (lane & 15);
  int kb = kc + ((lane >> 4) & 1) * 8;
  const unsigned short* p = base + m * pitch + kb;
  FragU f;
  f.h[0] = *(const u16x8*)(p);
  f.h[1] = *(const u16x8*)(p + 16);
  return f.v;
}

// B fragment (32x16): element (k,n) at base[n*pitch + k] (i.e. B^T row-major);
// per-lane column n, K pattern {kc+kb+0..7, kc+kb+16..23}, kb = 8*(lane>=16).
__device__ __forceinline__ v16bf load_b_frag(const unsigned short* base, int pitch,
                                             int n0, int kc, int lane) {
  int n = n0 + (lane & 15);
  int kb = kc + ((lane >> 4) & 1) * 8;
  const unsigned short* p = base + n * pitch + kb;
  FragU f;
  f.h[0] = *(const u16x8*)(p);
  f.h[1] = *(const u16x8*)(p + 16);
  return f.v;
}

// ---------------- Tensor Data Mover (TDM) 2D tile load ----------------
__device__ __forceinline__ unsigned int lds_addr32(const void* p) {
  return (unsigned int)(unsigned long long)p;  // low 32 bits = LDS byte offset
}

__device__ __forceinline__ void tdm_load_2d(unsigned int lds_off, const void* gptr,
                                            unsigned int tile_d0, unsigned int tile_d1,
                                            unsigned int tensor_d0, unsigned int tensor_d1,
                                            unsigned int d0_stride,
                                            unsigned int pad_int_code,
                                            unsigned int pad_amt_code) {
  unsigned long long ga = (unsigned long long)gptr;
  u32x4 g0;
  g0[0] = 1u;  // count=1, user descriptor
  g0[1] = lds_off;
  g0[2] = (unsigned int)(ga & 0xffffffffu);
  g0[3] = (unsigned int)((ga >> 32) & 0x01ffffffu) | (2u << 30);  // type=2
  i32x8 g1;
  unsigned int w0 = (1u << 16)  // data_size = 2 bytes
                  | (1u << 20)  // pad_enable
                  | (pad_int_code << 22) | (pad_amt_code << 25);
  g1[0] = (int)w0;
  g1[1] = (int)((tensor_d0 & 0xffffu) << 16);
  g1[2] = (int)(((tensor_d0 >> 16) & 0xffffu) | ((tensor_d1 & 0xffffu) << 16));
  g1[3] = (int)(((tensor_d1 >> 16) & 0xffffu) | ((tile_d0 & 0xffffu) << 16));
  g1[4] = (int)(tile_d1 & 0xffffu);
  g1[5] = (int)d0_stride;
  g1[6] = 0;
  g1[7] = 0;
  asm volatile("tensor_load_to_lds %0, %1" : : "s"(g0), "s"(g1) : "memory");
}

__device__ __forceinline__ void tdm_wait0() {
  __builtin_amdgcn_s_wait_tensorcnt(0);
}
__device__ __forceinline__ void tdm_wait2() {
  __builtin_amdgcn_s_wait_tensorcnt(2);
}

// ---------------- one-time precision conversion kernels ----------------
__global__ void cast_f32_bf16_kernel(const float* __restrict__ in,
                                     unsigned short* __restrict__ out, long n) {
  long i = (long)blockIdx.x * blockDim.x + threadIdx.x;
  long stride = (long)gridDim.x * blockDim.x;
  for (; i < n; i += stride) out[i] = f2bf(in[i]);
}

// W [K,N] f32 -> WT [N,K] bf16 (tiled transpose)
__global__ __launch_bounds__(256) void transpose_cast_kernel(
    const float* __restrict__ W, unsigned short* __restrict__ WT, int K, int N) {
  __shared__ float tile[32][33];
  int n0 = blockIdx.x * 32;
  int k0 = blockIdx.y * 32;
  int tx = threadIdx.x & 31, ty = threadIdx.x >> 5;  // 32 x 8
  for (int i = ty; i < 32; i += 8)
    tile[i][tx] = W[(size_t)(k0 + i) * N + n0 + tx];
  __syncthreads();
  for (int i = ty; i < 32; i += 8)
    WT[(size_t)(n0 + i) * K + k0 + tx] = f2bf(tile[tx][i]);
}

// ---------------- GEMM (+ fused RoPE epilogue), TDM double-buffered --------
// C[M,N] = A[M,K] * BT[N,K]^T (both bf16 row-major), f32 WMMA accumulate.
// mode 0: RoPE+scale -> bf16 q [B,H,T,HD]
// mode 1: RoPE -> f32 k-cache + bf16 copy [B,KV,T,HD]
// mode 2: plain -> f32 v-cache [B,KV,T,HD] + bf16 TRANSPOSED copy [B,KV,HD,T]
// mode 3: plain -> f32 row-major [M,N]
#define TM 128
#define TN 128
#define KC2 64
#define AP 72    // LDS pitch for A / BT tiles (KC2 + 8)
#define LDCP 132

union GemmLDS {
  struct { unsigned short a[2][TM * AP]; unsigned short bt[2][TN * AP]; } s;
  float c[64 * LDCP];
};

__global__ __launch_bounds__(256) void gemm_rope_kernel(
    const unsigned short* __restrict__ Abf, const unsigned short* __restrict__ BT,
    int K, int N, const float* __restrict__ cosp, const float* __restrict__ sinp,
    float* __restrict__ dstF, unsigned short* __restrict__ dstB, int mode) {
  __shared__ __align__(16) GemmLDS lds;

  const int tid = threadIdx.x;
  const int lane = tid & 31;
  const int w = tid >> 5;
  const int wm = w & 3;   // 4 wave-rows of 32
  const int wn = w >> 2;  // 2 wave-cols of 64
  const int hi8 = ((lane >> 4) & 1) * 8;
  const int m0 = blockIdx.x * TM;
  const int n0 = blockIdx.y * TN;

  v8f acc[2][4];
#pragma unroll
  for (int i = 0; i < 2; ++i)
#pragma unroll
    for (int j = 0; j < 4; ++j)
      acc[i][j] = v8f{0.f, 0.f, 0.f, 0.f, 0.f, 0.f, 0.f, 0.f};

  // preload first K-chunk into buffer 0 (rows of 64 bf16 = 32 DW -> pad code 4/3)
  if (w == 0) {
    tdm_load_2d(lds_addr32(lds.s.a[0]), Abf + (size_t)m0 * K,
                KC2, TM, (unsigned)K, (unsigned)MTOT, (unsigned)K, 4, 3);
    tdm_load_2d(lds_addr32(lds.s.bt[0]), BT + (size_t)n0 * K,
                KC2, TN, (unsigned)K, (unsigned)N, (unsigned)K, 4, 3);
  }

  int p = 0;
  for (int kk = 0; kk < K; kk += KC2) {
    if (w == 0) {
      if (kk + KC2 < K) {  // prefetch next chunk into the other buffer
        tdm_load_2d(lds_addr32(lds.s.a[p ^ 1]), Abf + (size_t)m0 * K + kk + KC2,
                    KC2, TM, (unsigned)K, (unsigned)MTOT, (unsigned)K, 4, 3);
        tdm_load_2d(lds_addr32(lds.s.bt[p ^ 1]), BT + (size_t)n0 * K + kk + KC2,
                    KC2, TN, (unsigned)K, (unsigned)N, (unsigned)K, 4, 3);
        tdm_wait2();  // in-order: <=2 outstanding means buffer p has landed
      } else {
        tdm_wait0();
      }
    }
    __syncthreads();  // buffer p visible to all waves
#pragma unroll
    for (int ks = 0; ks < 2; ++ks) {
      v16bf af[2], bfm[4];
#pragma unroll
      for (int i = 0; i < 2; ++i)
        af[i] = load_a_frag(lds.s.a[p], AP, wm * 32 + i * 16, ks * 32, lane);
#pragma unroll
      for (int j = 0; j < 4; ++j)
        bfm[j] = load_b_frag(lds.s.bt[p], AP, wn * 64 + j * 16, ks * 32, lane);
#pragma unroll
      for (int i = 0; i < 2; ++i)
#pragma unroll
        for (int j = 0; j < 4; ++j)
          acc[i][j] = wmma_bf16(af[i], bfm[j], acc[i][j]);
    }
    __syncthreads();  // all reads of buffer p done; it may be overwritten
    p ^= 1;
  }

  const float SCALE = 0.08838834764831845f;  // HD^-0.5
#pragma unroll
  for (int half = 0; half < 2; ++half) {
    __syncthreads();
    if ((wm >> 1) == half) {
#pragma unroll
      for (int i = 0; i < 2; ++i)
#pragma unroll
        for (int j = 0; j < 4; ++j)
#pragma unroll
          for (int r = 0; r < 8; ++r) {
            int ml = wm * 32 + i * 16 + r + hi8 - half * 64;
            int c = wn * 64 + j * 16 + (lane & 15);
            lds.c[ml * LDCP + c] = acc[i][j][r];
          }
    }
    __syncthreads();
    for (int e = tid; e < 64 * TN; e += 256) {
      int r = e >> 7;
      int c = e & (TN - 1);
      int m = m0 + half * 64 + r;
      float val = lds.c[r * LDCP + c];
      if (mode == 3) {
        dstF[(size_t)m * N + (n0 + c)] = val;
        continue;
      }
      int b = m >> 11, t = m & (T_ - 1);
      int h = (n0 + c) >> 7, d = c;  // N tiles are HD-aligned
      if (mode <= 1) {  // RoPE
        float other = lds.c[r * LDCP + (c ^ 64)];
        float cv = cosp[t * HD_ + d], sv = sinp[t * HD_ + d];
        val = cv * val + sv * ((d < 64) ? -other : other);
      }
      if (mode == 0) {
        size_t idx = (((size_t)b * H_ + h) * T_ + t) * HD_ + d;
        dstB[idx] = f2bf(val * SCALE);
      } else {
        size_t idxF = (((size_t)b * KV_ + h) * T_ + t) * HD_ + d;
        dstF[idxF] = val;
        size_t idxB = (mode == 2)
                          ? ((((size_t)b * KV_ + h) * HD_ + d) * T_ + t)  // V^T
                          : idxF;
        dstB[idxB] = f2bf(val);
      }
    }
  }
}

// ---------------- Flash attention (causal, GQA), TDM pipelined -------------
#define FM 64   // q rows per block
#define FK 64   // k cols per k-block
#define QP 136  // bf16 pitch for Q / K tiles (HD_+8)
#define VP 72   // bf16 pitch for V^T tile (FK+8)
#define PP 72   // bf16 pitch for P tile
#define SP 68   // f32 pitch for S tile (FK+4)

union KSU {
  unsigned short kk[FK * QP];  // K tile, row-major [k-time][d]
  float ss[FM * SP];           // S scores (same 17408B footprint)
};

__global__ __launch_bounds__(256) void flash_attn_kernel(
    const unsigned short* __restrict__ qb, const unsigned short* __restrict__ kb,
    const unsigned short* __restrict__ vtb, unsigned short* __restrict__ yb) {
  __shared__ __align__(16) unsigned short sQ[FM * QP];
  __shared__ __align__(16) KSU sKS;
  __shared__ __align__(16) unsigned short sVT[2][HD_ * VP];  // double buffered
  __shared__ __align__(16) unsigned short sP[FM * PP];
  __shared__ float sM[FM], sL[FM], sAl[FM];

  const int tid = threadIdx.x;
  const int lane = tid & 31;
  const int w = tid >> 5;
  const int hi8 = ((lane >> 4) & 1) * 8;

  const int nqt = T_ / FM;  // 32
  int bid = blockIdx.x;
  int qi = bid % nqt;
  int h = (bid / nqt) % H_;
  int b = bid / (nqt * H_);
  int g = h / REP_;  // kv head
  int qt0 = qi * FM;

  const unsigned short* qsrc = qb + (((size_t)b * H_ + h) * T_) * HD_;
  const unsigned short* ksrc = kb + (((size_t)b * KV_ + g) * T_) * HD_;
  const unsigned short* vtsrc = vtb + (((size_t)b * KV_ + g) * HD_) * T_;  // [HD][T]

  if (w == 0) {
    // Q tile + first K / V^T tiles (rows 128 bf16 = 64 DW -> 5/3; 64 bf16 -> 4/3)
    tdm_load_2d(lds_addr32(sQ), qsrc + (size_t)qt0 * HD_, HD_, FM, HD_, T_, HD_, 5, 3);
    tdm_load_2d(lds_addr32(sKS.kk), ksrc, HD_, FK, HD_, T_, HD_, 5, 3);
    tdm_load_2d(lds_addr32(sVT[0]), vtsrc, FK, HD_, T_, HD_, T_, 4, 3);
  }
  if (tid < FM) { sM[tid] = -1e30f; sL[tid] = 0.0f; }

  const int wmS = w & 3, wnS = w >> 2;  // S: 16 rows x 32 cols per wave
  const int wmO = w & 3, wnO = w >> 2;  // O: 16 rows x 64 d per wave

  v8f oacc[4];
#pragma unroll
  for (int j = 0; j < 4; ++j) oacc[j] = v8f{0.f, 0.f, 0.f, 0.f, 0.f, 0.f, 0.f, 0.f};

  int p = 0;
  const int nkb = qt0 / FK + 1;  // causal
  for (int kbi = 0; kbi < nkb; ++kbi) {
    const int kt0 = kbi * FK;
    if (w == 0) tdm_wait0();  // this iteration's K / V^T (and Q) have landed
    __syncthreads();

    // S = Q * K^T (K tile rows are B^T columns directly)
    v8f s[2];
    s[0] = v8f{0.f, 0.f, 0.f, 0.f, 0.f, 0.f, 0.f, 0.f};
    s[1] = v8f{0.f, 0.f, 0.f, 0.f, 0.f, 0.f, 0.f, 0.f};
#pragma unroll
    for (int ks = 0; ks < 4; ++ks) {
      v16bf a = load_a_frag(sQ, QP, wmS * 16, ks * 32, lane);
#pragma unroll
      for (int j = 0; j < 2; ++j) {
        v16bf bfr = load_b_frag(sKS.kk, QP, wnS * 32 + j * 16, ks * 32, lane);
        s[j] = wmma_bf16(a, bfr, s[j]);
      }
    }
    __syncthreads();  // K-tile reads done before overlaying scores
#pragma unroll
    for (int j = 0; j < 2; ++j)
#pragma unroll
      for (int r = 0; r < 8; ++r) {
        int row = wmS * 16 + r + hi8;
        int col = wnS * 32 + j * 16 + (lane & 15);
        sKS.ss[row * SP + col] = s[j][r];
      }
    __syncthreads();

    // online softmax: one thread per q row
    if (tid < FM) {
      int r = tid, tq = qt0 + r;
      float mprev = sM[r], rowmax = mprev;
      for (int c = 0; c < FK; ++c)
        if (kt0 + c <= tq) rowmax = fmaxf(rowmax, sKS.ss[r * SP + c]);
      float alpha = __expf(mprev - rowmax);
      float sum = 0.0f;
      for (int c = 0; c < FK; ++c) {
        float p2 = (kt0 + c <= tq) ? __expf(sKS.ss[r * SP + c] - rowmax) : 0.0f;
        sP[r * PP + c] = f2bf(p2);
        sum += p2;
      }
      sL[r] = sL[r] * alpha + sum;
      sM[r] = rowmax;
      sAl[r] = alpha;
    }
    __syncthreads();

    // scores consumed: prefetch next K / V^T while we do P*V
    if (w == 0 && kbi + 1 < nkb) {
      tdm_load_2d(lds_addr32(sKS.kk), ksrc + (size_t)(kt0 + FK) * HD_,
                  HD_, FK, HD_, T_, HD_, 5, 3);
      tdm_load_2d(lds_addr32(sVT[p ^ 1]), vtsrc + (kt0 + FK),
                  FK, HD_, T_, HD_, T_, 4, 3);
    }

    // O = alpha*O + P*V
#pragma unroll
    for (int j = 0; j < 4; ++j)
#pragma unroll
      for (int r = 0; r < 8; ++r)
        oacc[j][r] = oacc[j][r] * sAl[wmO * 16 + r + hi8];
#pragma unroll
    for (int ks = 0; ks < 2; ++ks) {
      v16bf a = load_a_frag(sP, PP, wmO * 16, ks * 32, lane);
#pragma unroll
      for (int j = 0; j < 4; ++j) {
        v16bf bfr = load_b_frag(sVT[p], VP, wnO * 64 + j * 16, ks * 32, lane);
        oacc[j] = wmma_bf16(a, bfr, oacc[j]);
      }
    }
    p ^= 1;
  }
  __syncthreads();
#pragma unroll
  for (int j = 0; j < 4; ++j)
#pragma unroll
    for (int r = 0; r < 8; ++r) {
      int row = wmO * 16 + r + hi8;
      int tq = qt0 + row;
      int d = wnO * 64 + j * 16 + (lane & 15);
      float v = oacc[j][r] / sL[row];
      yb[((size_t)(b * T_) + tq) * (H_ * HD_) + h * HD_ + d] = f2bf(v);
    }
}

extern "C" void kernel_launch(void* const* d_in, const int* in_sizes, int n_in,
                              void* d_out, int out_size, void* d_ws, size_t ws_size,
                              hipStream_t stream) {
  const float* x = (const float*)d_in[0];
  const float* cosp = (const float*)d_in[1];
  const float* sinp = (const float*)d_in[2];
  const float* Wq = (const float*)d_in[3];
  const float* Wk = (const float*)d_in[4];
  const float* Wv = (const float*)d_in[5];
  const float* Wo = (const float*)d_in[6];
  float* out = (float*)d_out;

  char* ws = (char*)d_ws;
  size_t off = 0;
  auto alloc = [&](size_t elems) {
    unsigned short* p = (unsigned short*)(ws + off);
    off += elems * sizeof(unsigned short);
    return p;
  };
  unsigned short* xb = alloc((size_t)MTOT * E_);            // x bf16 [M,E]
  unsigned short* wqt = alloc((size_t)(H_ * HD_) * E_);     // Wq^T [N,K]
  unsigned short* wkt = alloc((size_t)(KV_ * HD_) * E_);    // Wk^T
  unsigned short* wvt = alloc((size_t)(KV_ * HD_) * E_);    // Wv^T
  unsigned short* wot = alloc((size_t)E_ * (H_ * HD_));     // Wo^T
  unsigned short* qbf = alloc((size_t)B_ * H_ * T_ * HD_);  // q rope'd bf16
  unsigned short* kbf = alloc((size_t)B_ * KV_ * T_ * HD_); // k rope'd bf16
  unsigned short* vbt = alloc((size_t)B_ * KV_ * HD_ * T_); // v bf16 TRANSPOSED
  unsigned short* yb = alloc((size_t)MTOT * (H_ * HD_));    // attention out bf16

  cast_f32_bf16_kernel<<<1024, 256, 0, stream>>>(x, xb, (long)MTOT * E_);
  transpose_cast_kernel<<<dim3((H_ * HD_) / 32, E_ / 32), 256, 0, stream>>>(
      Wq, wqt, E_, H_ * HD_);
  transpose_cast_kernel<<<dim3((KV_ * HD_) / 32, E_ / 32), 256, 0, stream>>>(
      Wk, wkt, E_, KV_ * HD_);
  transpose_cast_kernel<<<dim3((KV_ * HD_) / 32, E_ / 32), 256, 0, stream>>>(
      Wv, wvt, E_, KV_ * HD_);
  transpose_cast_kernel<<<dim3(E_ / 32, (H_ * HD_) / 32), 256, 0, stream>>>(
      Wo, wot, H_ * HD_, E_);

  float* kout = out + (size_t)MTOT * E_;             // k cache section
  float* vout = kout + (size_t)B_ * KV_ * T_ * HD_;  // v cache section

  dim3 blk(256);
  gemm_rope_kernel<<<dim3(MTOT / TM, (H_ * HD_) / TN), blk, 0, stream>>>(
      xb, wqt, E_, H_ * HD_, cosp, sinp, nullptr, qbf, 0);
  gemm_rope_kernel<<<dim3(MTOT / TM, (KV_ * HD_) / TN), blk, 0, stream>>>(
      xb, wkt, E_, KV_ * HD_, cosp, sinp, kout, kbf, 1);
  gemm_rope_kernel<<<dim3(MTOT / TM, (KV_ * HD_) / TN), blk, 0, stream>>>(
      xb, wvt, E_, KV_ * HD_, cosp, sinp, vout, vbt, 2);
  flash_attn_kernel<<<dim3(B_ * H_ * (T_ / FM)), blk, 0, stream>>>(qbf, kbf, vbt, yb);
  gemm_rope_kernel<<<dim3(MTOT / TM, E_ / TN), blk, 0, stream>>>(
      yb, wot, H_ * HD_, E_, nullptr, nullptr, out, nullptr, 3);
}
